// MultiHeadAttention_51402168598915
// MI455X (gfx1250) — matrix operand order repaired and checked by reading, not compiled, not run on hardware
//
#include <hip/hip_runtime.h>
#include <hip/hip_bf16.h>
#include <stdint.h>

// ---------------------------------------------------------------------------
// MHA block for MI455X (gfx1250, wave32, WMMA + double-buffered async-to-LDS).
// B=8, S=1024, D=1024, H=16, DK=DV=64.
// ---------------------------------------------------------------------------

typedef _Float16 half_t;
typedef __attribute__((ext_vector_type(16))) _Float16 v16h;
typedef __attribute__((ext_vector_type(8)))  _Float16 v8h;
typedef __attribute__((ext_vector_type(4)))  _Float16 v4h;
typedef __attribute__((ext_vector_type(8)))  float    v8f;

#define BB 8
#define SS 1024
#define DD 1024
#define HH 16
#define DKV 64

__device__ __forceinline__ v16h cat16(v8h lo, v8h hi) {
  return __builtin_shufflevector(lo, hi, 0,1,2,3,4,5,6,7,8,9,10,11,12,13,14,15);
}

__device__ __forceinline__ v8f wmma_f16(v16h a, v16h b, v8f c) {
  return __builtin_amdgcn_wmma_f32_16x16x32_f16(false, a, false, b, (short)0, c,
                                                false, false);
}

// CDNA5 async global->LDS copy (16B per lane), tracked with ASYNCcnt.
__device__ __forceinline__ void async_load_b128(uint32_t lds_byte_addr, const void* gaddr) {
  asm volatile("global_load_async_to_lds_b128 %0, %1, off"
               :: "v"(lds_byte_addr), "v"((uint64_t)(uintptr_t)gaddr)
               : "memory");
}
// Async loads complete in order: waiting <=N leaves the newest N in flight.
__device__ __forceinline__ void wait_async0() {
  asm volatile("s_wait_asynccnt 0x0" ::: "memory");
}
__device__ __forceinline__ void wait_async6() {
  asm volatile("s_wait_asynccnt 0x6" ::: "memory");
}
__device__ __forceinline__ void wait_async8() {
  asm volatile("s_wait_asynccnt 0x8" ::: "memory");
}
__device__ __forceinline__ uint32_t lds_addr(const void* p) {
  return (uint32_t)(uintptr_t)p;
}

// ---------------------------------------------------------------------------
// f32 -> f16 elementwise convert (vectorized by 4)
// ---------------------------------------------------------------------------
__global__ __launch_bounds__(256) void cvt16_kernel(const float* __restrict__ in,
                                                    half_t* __restrict__ out, int n4) {
  int i = blockIdx.x * 256 + threadIdx.x;
  if (i < n4) {
    float4 f = ((const float4*)in)[i];
    v4h o;
    o[0] = (half_t)f.x; o[1] = (half_t)f.y; o[2] = (half_t)f.z; o[3] = (half_t)f.w;
    ((v4h*)out)[i] = o;
  }
}

// ---------------------------------------------------------------------------
// Weight transpose + convert: Wt[n][k] = (f16) W[k][n], 1024x1024
// ---------------------------------------------------------------------------
__global__ __launch_bounds__(256) void cvt_transpose_kernel(const float* __restrict__ W,
                                                            half_t* __restrict__ Wt) {
  __shared__ float tile[32][33];
  int k0 = blockIdx.x * 32, n0 = blockIdx.y * 32;
  int tx = threadIdx.x & 31, ty = threadIdx.x >> 5;  // ty: 0..7
  for (int i = 0; i < 32; i += 8)
    tile[ty + i][tx] = W[(size_t)(k0 + ty + i) * DD + n0 + tx];
  __syncthreads();
  for (int i = 0; i < 32; i += 8)
    Wt[(size_t)(n0 + ty + i) * DD + k0 + tx] = (half_t)tile[tx][ty + i];
}

// ---------------------------------------------------------------------------
// WMMA GEMM: Out = A[M x 1024] * W[1024 x 1024]  (W given transposed: Bt[n][k])
// Block: 128 threads (4 waves), tile 128(M) x 64(N).
// Wave w owns rows w*32..+32 (2 A-frag sets) x all 64 N columns: 8 WMMA/K-step.
// Double-buffered async global->LDS staging (6 async b128 per thread per tile).
// mode 0: f16 scatter to [B][H][S][64]; mode 1: f16 scatter to [B][H][64][S];
// mode 2: f32 = acc + residual, row-major [M][1024].
// ---------------------------------------------------------------------------
__global__ __launch_bounds__(128) void gemm_wmma_kernel(
    const half_t* __restrict__ A, const half_t* __restrict__ Bt,
    half_t* __restrict__ outH, float* __restrict__ outF,
    const float* __restrict__ resid, int mode) {
  const int K = DD;
  __shared__ half_t Asub[2][128][40];   // pitch 80 B
  __shared__ half_t Bsub[2][64][40];

  int tid  = threadIdx.x;
  int wave = tid >> 5, lane = tid & 31;
  int l15  = lane & 15, hi = lane >> 4;
  int n0 = blockIdx.x * 64;
  int m0 = blockIdx.y * 128;

  // Staging: thread handles A rows (tid>>2)+{0,32,64,96}, B rows (tid>>2)+{0,32},
  // column chunk (tid&3)*8 halves; one b128 async copy each.
  int srow = tid >> 2, scol = (tid & 3) * 8;
  const half_t* ag = A  + (size_t)(m0 + srow) * K + scol;
  const half_t* bg = Bt + (size_t)(n0 + srow) * K + scol;
  uint32_t aL[2] = { lds_addr(&Asub[0][srow][scol]), lds_addr(&Asub[1][srow][scol]) };
  uint32_t bL[2] = { lds_addr(&Bsub[0][srow][scol]), lds_addr(&Bsub[1][srow][scol]) };

  auto stage = [&](int k0, int buf) {
#pragma unroll
    for (int i = 0; i < 4; ++i)
      async_load_b128(aL[buf] + i * 32 * 80, ag + (size_t)i * 32 * K + k0);
#pragma unroll
    for (int i = 0; i < 2; ++i)
      async_load_b128(bL[buf] + i * 32 * 80, bg + (size_t)i * 32 * K + k0);
    __builtin_prefetch(ag + k0 + 32, 0, 1);
    __builtin_prefetch(bg + k0 + 32, 0, 1);
  };

  v8f acc[2][4] = {};

  stage(0, 0);
  for (int k0 = 0; k0 < K; k0 += 32) {
    int buf = (k0 >> 5) & 1;
    if (k0 + 32 < K) { stage(k0 + 32, buf ^ 1); wait_async6(); }
    else             { wait_async0(); }
    __syncthreads();  // this tile visible from all waves

    int am = wave * 32 + l15;
    v16h a0 = cat16(*(const v8h*)&Asub[buf][am][hi * 8],
                    *(const v8h*)&Asub[buf][am][16 + hi * 8]);
    v16h a1 = cat16(*(const v8h*)&Asub[buf][am + 16][hi * 8],
                    *(const v8h*)&Asub[buf][am + 16][16 + hi * 8]);
    v16h bf[4];
#pragma unroll
    for (int ns = 0; ns < 4; ++ns) {
      int bn = ns * 16 + l15;
      bf[ns] = cat16(*(const v8h*)&Bsub[buf][bn][hi * 8],
                     *(const v8h*)&Bsub[buf][bn][16 + hi * 8]);
    }
#pragma unroll
    for (int ns = 0; ns < 4; ++ns) {
      acc[0][ns] = wmma_f16(a0, bf[ns], acc[0][ns]);
      acc[1][ns] = wmma_f16(a1, bf[ns], acc[1][ns]);
    }
    __syncthreads();  // buffer free before it is restaged
  }

  if (mode == 2) {
#pragma unroll
    for (int aset = 0; aset < 2; ++aset)
      for (int ns = 0; ns < 4; ++ns) {
        int n = n0 + ns * 16 + l15;
        for (int r = 0; r < 8; ++r) {
          int m = m0 + wave * 32 + aset * 16 + hi * 8 + r;
          size_t idx = (size_t)m * DD + n;
          outF[idx] = acc[aset][ns][r] + resid[idx];
        }
      }
  } else {
#pragma unroll
    for (int aset = 0; aset < 2; ++aset)
      for (int ns = 0; ns < 4; ++ns) {
        int n = n0 + ns * 16 + l15;
        int h = n >> 6, d = n & 63;
        for (int r = 0; r < 8; ++r) {
          int m = m0 + wave * 32 + aset * 16 + hi * 8 + r;
          int b = m >> 10, s = m & 1023;
          size_t idx;
          if (mode == 0) idx = ((size_t)(b * HH + h) * SS + s) * DKV + d;  // [B][H][S][64]
          else           idx = ((size_t)(b * HH + h) * DKV + d) * SS + s;  // [B][H][64][S]
          outH[idx] = (half_t)acc[aset][ns][r];
        }
      }
  }
}

// ---------------------------------------------------------------------------
// Flash attention: grid (S/64, B*H), 128 threads (4 waves).
// Wave w owns query rows q0+w*16 .. +16. Key tiles of 64, online softmax.
// Double-buffered async staging of K / V^T tiles (8 async b128 per thread).
// ---------------------------------------------------------------------------
#define SD 72
__global__ __launch_bounds__(128) void attn_wmma_kernel(
    const half_t* __restrict__ Qh, const half_t* __restrict__ Kh,
    const half_t* __restrict__ Vt, const int* __restrict__ mask,
    half_t* __restrict__ att) {
  __shared__ half_t Kt[2][64][SD];      // [key][dk]
  __shared__ half_t Vtile[2][64][SD];   // [dv][key]
  __shared__ half_t Pt[4][16][SD];      // per-wave probability tile [q][key]

  int tid = threadIdx.x, wave = tid >> 5, lane = tid & 31;
  int l15 = lane & 15, hi = lane >> 4;
  int bh = blockIdx.y;               // b*16 + h
  int b  = bh >> 4, h = bh & 15;
  int q0 = blockIdx.x * 64;

  const half_t* Qbase = Qh + (size_t)bh * SS * DKV;
  const half_t* Kbase = Kh + (size_t)bh * SS * DKV;
  const half_t* Vbase = Vt + (size_t)bh * DKV * SS;

  // Q fragments for this wave's 16 rows (DK=64 -> two K=32 fragments).
  int qm = q0 + wave * 16 + l15;
  const half_t* qrow = Qbase + (size_t)qm * DKV;
  v16h qa0 = cat16(*(const v8h*)&qrow[hi * 8],      *(const v8h*)&qrow[16 + hi * 8]);
  v16h qa1 = cat16(*(const v8h*)&qrow[32 + hi * 8], *(const v8h*)&qrow[48 + hi * 8]);

  // Staging: thread handles row tid>>1, 32-half segment (tid&1)*32.
  int sr = tid >> 1, sg = (tid & 1) * 32;
  uint32_t ktL[2] = { lds_addr(&Kt[0][sr][sg]), lds_addr(&Kt[1][sr][sg]) };
  uint32_t vtL[2] = { lds_addr(&Vtile[0][sr][sg]), lds_addr(&Vtile[1][sr][sg]) };

  auto stageKV = [&](int jt, int buf) {
    const half_t* ksrc = Kbase + (size_t)(jt + sr) * DKV + sg;
    const half_t* vsrc = Vbase + (size_t)sr * SS + jt + sg;
#pragma unroll
    for (int c = 0; c < 32; c += 8) {
      async_load_b128(ktL[buf] + c * 2, ksrc + c);
      async_load_b128(vtL[buf] + c * 2, vsrc + c);
    }
  };

  float mrow[8], lrow[8];
  v8f   Oacc[4] = {};
  for (int r = 0; r < 8; ++r) { mrow[r] = -__builtin_inff(); lrow[r] = 0.f; }

  stageKV(0, 0);
  for (int jt = 0; jt < SS; jt += 64) {
    int buf = (jt >> 6) & 1;
    if (jt + 64 < SS) { stageKV(jt + 64, buf ^ 1); wait_async8(); }
    else              { wait_async0(); }
    __syncthreads();

    // Scores: 4 sub-tiles of 16 keys; 2 WMMAs each (dk = 0..31, 32..63).
    v8f sc[4];
#pragma unroll
    for (int ns = 0; ns < 4; ++ns) {
      int kn = ns * 16 + l15;
      v16h bk0 = cat16(*(const v8h*)&Kt[buf][kn][hi * 8],
                       *(const v8h*)&Kt[buf][kn][16 + hi * 8]);
      v16h bk1 = cat16(*(const v8h*)&Kt[buf][kn][32 + hi * 8],
                       *(const v8h*)&Kt[buf][kn][48 + hi * 8]);
      v8f c = {};
      c = wmma_f16(qa0, bk0, c);
      c = wmma_f16(qa1, bk1, c);
      for (int r = 0; r < 8; ++r) {
        int qg = q0 + wave * 16 + hi * 8 + r;
        int kg = jt + ns * 16 + l15;
        float sv = c[r] * 0.125f;  // 1/sqrt(64)
        int mk = mask[((size_t)b * SS + qg) * SS + kg];
        sc[ns][r] = (mk == 0) ? -1e9f : sv;
      }
    }

    // Online softmax per row (rows live in 16-lane groups; xor<=8 stays inside).
    float fscale[8];
#pragma unroll
    for (int r = 0; r < 8; ++r) {
      float mx = fmaxf(fmaxf(sc[0][r], sc[1][r]), fmaxf(sc[2][r], sc[3][r]));
      for (int off = 8; off >= 1; off >>= 1) mx = fmaxf(mx, __shfl_xor(mx, off, 32));
      float mnew = fmaxf(mrow[r], mx);
      fscale[r] = __expf(mrow[r] - mnew);
      mrow[r] = mnew;
      float psum = 0.f;
      for (int ns = 0; ns < 4; ++ns) {
        float p = __expf(sc[ns][r] - mnew);
        psum += p;
        Pt[wave][hi * 8 + r][ns * 16 + l15] = (half_t)p;
      }
      for (int off = 8; off >= 1; off >>= 1) psum += __shfl_xor(psum, off, 32);
      lrow[r] = lrow[r] * fscale[r] + psum;
    }
#pragma unroll
    for (int nv = 0; nv < 4; ++nv)
      for (int r = 0; r < 8; ++r) Oacc[nv][r] *= fscale[r];

    // P @ V: P as A-fragments (keys 0..31, 32..63) from wave-private LDS.
    v16h pa0 = cat16(*(const v8h*)&Pt[wave][l15][hi * 8],
                     *(const v8h*)&Pt[wave][l15][16 + hi * 8]);
    v16h pa1 = cat16(*(const v8h*)&Pt[wave][l15][32 + hi * 8],
                     *(const v8h*)&Pt[wave][l15][48 + hi * 8]);
#pragma unroll
    for (int nv = 0; nv < 4; ++nv) {
      int dv = nv * 16 + l15;
      v16h bv0 = cat16(*(const v8h*)&Vtile[buf][dv][hi * 8],
                       *(const v8h*)&Vtile[buf][dv][16 + hi * 8]);
      v16h bv1 = cat16(*(const v8h*)&Vtile[buf][dv][32 + hi * 8],
                       *(const v8h*)&Vtile[buf][dv][48 + hi * 8]);
      Oacc[nv] = wmma_f16(pa0, bv0, Oacc[nv]);
      Oacc[nv] = wmma_f16(pa1, bv1, Oacc[nv]);
    }
    __syncthreads();  // buffer free before it is restaged
  }

  // Normalize and write [B*S][H*64] f16.
#pragma unroll
  for (int r = 0; r < 8; ++r) {
    float rinv = (lrow[r] > 0.f) ? 1.f / lrow[r] : 0.f;
    int qg = q0 + wave * 16 + hi * 8 + r;
    size_t rowoff = ((size_t)b * SS + qg) * DD + h * DKV;
    for (int nv = 0; nv < 4; ++nv)
      att[rowoff + nv * 16 + l15] = (half_t)(Oacc[nv][r] * rinv);
  }
}

// ---------------------------------------------------------------------------
// LayerNorm over D=1024, one block (256 threads) per row.
// ---------------------------------------------------------------------------
__global__ __launch_bounds__(256) void layernorm_kernel(
    const float* __restrict__ x, const float* __restrict__ gamma,
    const float* __restrict__ beta, float* __restrict__ out) {
  int row = blockIdx.x;
  const float* xr = x + (size_t)row * DD;
  float v[4], s = 0.f, ss = 0.f;
  for (int i = 0; i < 4; ++i) {
    v[i] = xr[threadIdx.x + i * 256];
    s += v[i]; ss += v[i] * v[i];
  }
  for (int off = 16; off >= 1; off >>= 1) {
    s  += __shfl_xor(s,  off, 32);
    ss += __shfl_xor(ss, off, 32);
  }
  __shared__ float sbuf[8], ssbuf[8];
  int w = threadIdx.x >> 5;
  if ((threadIdx.x & 31) == 0) { sbuf[w] = s; ssbuf[w] = ss; }
  __syncthreads();
  float ts = 0.f, tss = 0.f;
  for (int i = 0; i < 8; ++i) { ts += sbuf[i]; tss += ssbuf[i]; }
  float mu  = ts * (1.f / DD);
  float var = tss * (1.f / DD) - mu * mu;
  float rstd = rsqrtf(var + 1e-6f);
  for (int i = 0; i < 4; ++i) {
    int c = threadIdx.x + i * 256;
    out[(size_t)row * DD + c] = (v[i] - mu) * rstd * gamma[c] + beta[c];
  }
}

// ---------------------------------------------------------------------------
extern "C" void kernel_launch(void* const* d_in, const int* in_sizes, int n_in,
                              void* d_out, int out_size, void* d_ws, size_t ws_size,
                              hipStream_t stream) {
  (void)in_sizes; (void)n_in; (void)out_size; (void)ws_size;
  const float* q     = (const float*)d_in[0];
  const float* k     = (const float*)d_in[1];
  const float* v     = (const float*)d_in[2];
  const int*   mask  = (const int*)d_in[3];
  const float* w_q   = (const float*)d_in[4];
  const float* w_k   = (const float*)d_in[5];
  const float* w_v   = (const float*)d_in[6];
  const float* w_fc  = (const float*)d_in[7];
  const float* gamma = (const float*)d_in[8];
  const float* beta  = (const float*)d_in[9];

  char* ws = (char*)d_ws;
  size_t o = 0;
  auto take = [&](size_t n) { char* p = ws + o; o += n; return p; };
  const size_t ACT16 = (size_t)BB * SS * DD * 2;   // 16 MiB
  const size_t W16   = (size_t)DD * DD * 2;        // 2 MiB
  half_t* q16   = (half_t*)take(ACT16);
  half_t* k16   = (half_t*)take(ACT16);
  half_t* v16   = (half_t*)take(ACT16);
  half_t* wqT   = (half_t*)take(W16);
  half_t* wkT   = (half_t*)take(W16);
  half_t* wvT   = (half_t*)take(W16);
  half_t* wfcT  = (half_t*)take(W16);
  half_t* Qh    = (half_t*)take(ACT16);
  half_t* Kh    = (half_t*)take(ACT16);
  half_t* Vth   = (half_t*)take(ACT16);
  half_t* att16 = (half_t*)take(ACT16);
  float*  preLN = (float*)take((size_t)BB * SS * DD * 4);

  // 1) Convert activations to f16.
  int n4 = BB * SS * DD / 4;
  cvt16_kernel<<<n4 / 256, 256, 0, stream>>>(q, q16, n4);
  cvt16_kernel<<<n4 / 256, 256, 0, stream>>>(k, k16, n4);
  cvt16_kernel<<<n4 / 256, 256, 0, stream>>>(v, v16, n4);

  // 2) Transpose-convert weights to [N][K] f16.
  dim3 tg(32, 32);
  cvt_transpose_kernel<<<tg, 256, 0, stream>>>(w_q,  wqT);
  cvt_transpose_kernel<<<tg, 256, 0, stream>>>(w_k,  wkT);
  cvt_transpose_kernel<<<tg, 256, 0, stream>>>(w_v,  wvT);
  cvt_transpose_kernel<<<tg, 256, 0, stream>>>(w_fc, wfcT);

  // 3) QKV projections (WMMA). Q/K -> [B][H][S][64], V -> [B][H][64][S].
  dim3 gg(DD / 64, BB * SS / 128);  // (16, 64)
  gemm_wmma_kernel<<<gg, 128, 0, stream>>>(q16, wqT, Qh,  nullptr, nullptr, 0);
  gemm_wmma_kernel<<<gg, 128, 0, stream>>>(k16, wkT, Kh,  nullptr, nullptr, 0);
  gemm_wmma_kernel<<<gg, 128, 0, stream>>>(v16, wvT, Vth, nullptr, nullptr, 1);

  // 4) Flash attention (WMMA scores + PV).
  dim3 ga(SS / 64, BB * HH);  // (16, 128)
  attn_wmma_kernel<<<ga, 128, 0, stream>>>(Qh, Kh, Vth, mask, att16);

  // 5) Output projection + residual (WMMA, f32 epilogue).
  gemm_wmma_kernel<<<gg, 128, 0, stream>>>(att16, wfcT, nullptr, preLN, q, 2);

  // 6) LayerNorm -> final f32 output.
  layernorm_kernel<<<BB * SS, 256, 0, stream>>>(preLN, gamma, beta, (float*)d_out);
}
